// ScaledDotProductAttention_46325517254937
// MI455X (gfx1250) — matrix operand order repaired and checked by reading, not compiled, not run on hardware
//
#include <hip/hip_runtime.h>

typedef _Float16 v16h __attribute__((ext_vector_type(16)));
typedef _Float16 v4h  __attribute__((ext_vector_type(4)));
typedef float    v8f  __attribute__((ext_vector_type(8)));

#define B_ 2
#define H_ 16
#define S_ 2048
#define D_ 64
#define WAVES 4
#define KT 32            // keys per tile
#define NKT (S_/KT)      // 64 tiles

// LDS row strides in f16 elements; 16B multiples, padded vs. bank conflicts
#define KSTR 72          // 144 B
#define VSTR 48          // 96 B
#define PSTR 40          // 80 B

#define LOG2E     1.44269504088896f
#define QSCALE    (0.125f * LOG2E)        // 1/sqrt(64) folded with log2(e)
#define MASKVAL   (-1e-9f * LOG2E)        // reference's -1e-9, in base-2 domain

__global__ __launch_bounds__(WAVES*32)
void fa_wmma_f16_kernel(const float* __restrict__ Q, const float* __restrict__ K,
                        const float* __restrict__ V, const int* __restrict__ pad,
                        float* __restrict__ Out)
{
  __shared__ _Float16 Klds[KT][KSTR];        // K tile, row-major [key][feat]
  __shared__ _Float16 Vt[D_][VSTR];          // V tile, transposed [feat][key]
  __shared__ _Float16 Plds[WAVES][16][PSTR]; // per-wave P transpose scratch
  __shared__ int      padk[S_];              // key pad mask for this batch

  const int bh    = blockIdx.y;       // b*H + h
  const int b     = bh / H_;
  const int tid   = threadIdx.x;
  const int wave  = tid >> 5;
  const int lane  = tid & 31;
  const int lh    = lane >> 4;        // lane-half (0/1)
  const int n     = lane & 15;
  const int qbase = blockIdx.x * (WAVES*16);
  const int q0    = qbase + wave*16;

  // last k-tile with any causally-visible key for this wave / this block
  const int diagkt  = (q0 + 15) >> 5;
  const int blockkt = (qbase + WAVES*16 - 1 + 15) >> 5;

  const float* Qb = Q   + (size_t)bh * S_ * D_;
  const float* Kb = K   + (size_t)bh * S_ * D_;
  const float* Vb = V   + (size_t)bh * S_ * D_;
  float*       Ob = Out + (size_t)bh * S_ * D_;

  for (int i = tid; i < S_; i += WAVES*32) padk[i] = pad[b*S_ + i];

  // ---- Q tile (16 rows) into WMMA A-layout, pre-scaled by log2e/sqrt(D) ----
  // A 16x32 f16: M = lane&15; h=0..15 -> k = 16*lh + h
  v16h aq[2];
  #pragma unroll
  for (int c = 0; c < 2; ++c) {
    #pragma unroll
    for (int v = 0; v < 8; ++v) {
      const int g = v >> 2, vv = v & 3;
      const int k = 32*c + 16*lh + 8*g + 2*vv;
      const float2 q2 = *(const float2*)&Qb[(size_t)(q0 + n) * D_ + k];
      aq[c][2*v]   = (_Float16)(q2.x * QSCALE);
      aq[c][2*v+1] = (_Float16)(q2.y * QSCALE);
    }
  }

  // all-ones matrix: rowsum(P) = P x ones ; colsum(V) = ones x V (layout-independent)
  v16h vones;
  #pragma unroll
  for (int h = 0; h < 16; ++h) vones[h] = (_Float16)1.0f;

  // per-lane query-row pad bits (C-layout rows M = r + 8*lh)
  int pq[8];
  #pragma unroll
  for (int r = 0; r < 8; ++r) pq[r] = pad[b*S_ + q0 + r + 8*lh];

  v8f acc[4], ctot[4];
  #pragma unroll
  for (int g = 0; g < 4; ++g)
    #pragma unroll
    for (int r = 0; r < 8; ++r) { acc[g][r] = 0.0f; ctot[g][r] = 0.0f; }

  float m_run[8], l_run[8];
  #pragma unroll
  for (int r = 0; r < 8; ++r) { m_run[r] = -3.0e38f; l_run[r] = 0.0f; }

  const v8f zero8 = {0.f,0.f,0.f,0.f,0.f,0.f,0.f,0.f};

  for (int kt = 0; kt < NKT; ++kt) {
    const int kbase  = kt * KT;
    const bool wantK = (kt <= blockkt);   // block-uniform: any wave still active?
    __syncthreads();  // previous tile's readers done (also fences padk on iter 0)

    // ---- cooperative stage: V tile always; K tile only while some wave active ----
    {
      const int key = tid >> 2;         // 0..31
      const int f0  = (tid & 3) * 16;   // 0,16,32,48
      #pragma unroll
      for (int j = 0; j < 4; ++j) {
        const float4 vv = *(const float4*)&Vb[(size_t)(kbase + key)*D_ + f0 + 4*j];
        Vt[f0+4*j+0][key] = (_Float16)vv.x;
        Vt[f0+4*j+1][key] = (_Float16)vv.y;
        Vt[f0+4*j+2][key] = (_Float16)vv.z;
        Vt[f0+4*j+3][key] = (_Float16)vv.w;
      }
      if (wantK) {
        #pragma unroll
        for (int j = 0; j < 4; ++j) {
          const float4 kk = *(const float4*)&Kb[(size_t)(kbase + key)*D_ + f0 + 4*j];
          v4h hk; hk[0]=(_Float16)kk.x; hk[1]=(_Float16)kk.y; hk[2]=(_Float16)kk.z; hk[3]=(_Float16)kk.w;
          __builtin_memcpy(&Klds[key][f0 + 4*j], &hk, 8);
        }
      }
    }

    if (kt + 1 < NKT) {  // prefetch next tile (global_prefetch_b8)
      __builtin_prefetch(&Vb[(size_t)(kbase + KT)*D_ + tid*16], 0, 0);
      if (kt + 1 <= blockkt)
        __builtin_prefetch(&Kb[(size_t)(kbase + KT)*D_ + tid*16], 0, 0);
    }

    __syncthreads();

    if (kt <= diagkt) {
      // ================= ACTIVE TILE: full flash-attention step =================
      // scores: S = Q.K^T (log2e-scaled) -> two 16x16 key groups, K=64 = 2x32
      v16h bk;
      v8f sc0 = zero8, sc1 = zero8;
      __builtin_memcpy(&bk, &Klds[n][16*lh], 32);
      sc0 = __builtin_amdgcn_wmma_f32_16x16x32_f16(false, aq[0], false, bk, (short)0, sc0, false, false);
      __builtin_memcpy(&bk, &Klds[n][32 + 16*lh], 32);
      sc0 = __builtin_amdgcn_wmma_f32_16x16x32_f16(false, aq[1], false, bk, (short)0, sc0, false, false);
      __builtin_memcpy(&bk, &Klds[16 + n][16*lh], 32);
      sc1 = __builtin_amdgcn_wmma_f32_16x16x32_f16(false, aq[0], false, bk, (short)0, sc1, false, false);
      __builtin_memcpy(&bk, &Klds[16 + n][32 + 16*lh], 32);
      sc1 = __builtin_amdgcn_wmma_f32_16x16x32_f16(false, aq[1], false, bk, (short)0, sc1, false, false);

      // faithful mask (-1e-9, NOT -inf: masked keys still feed softmax)
      const int key0 = kbase + n, key1 = kbase + 16 + n;
      const int pk0 = padk[key0], pk1 = padk[key1];
      float s0v[8], s1v[8], x[8];
      #pragma unroll
      for (int r = 0; r < 8; ++r) {
        const int qrow = q0 + r + 8*lh;
        float s0 = sc0[r], s1 = sc1[r];
        if (!(pq[r] && pk0 && key0 <= qrow)) s0 = MASKVAL;
        if (!(pq[r] && pk1 && key1 <= qrow)) s1 = MASKVAL;
        s0v[r] = s0; s1v[r] = s1;
        x[r] = fmaxf(s0, s1);
      }

      // row-max: batched 16-lane butterfly (8 independent bpermutes per step)
      #pragma unroll
      for (int step = 1; step <= 8; step <<= 1) {
        float y[8];
        #pragma unroll
        for (int r = 0; r < 8; ++r) y[r] = __shfl_xor(x[r], step, 32);
        #pragma unroll
        for (int r = 0; r < 8; ++r) x[r] = fmaxf(x[r], y[r]);
      }

      // online rescale + exp2, P -> LDS (C-layout -> A-layout transpose)
      #pragma unroll
      for (int r = 0; r < 8; ++r) {
        const float mn = fmaxf(m_run[r], x[r]);
        const float al = exp2f(m_run[r] - mn);
        const float p0 = exp2f(s0v[r] - mn);
        const float p1 = exp2f(s1v[r] - mn);
        l_run[r] *= al;           // tile row-sum added below via ones-WMMA
        m_run[r]  = mn;
        #pragma unroll
        for (int g = 0; g < 4; ++g) acc[g][r] *= al;
        Plds[wave][r + 8*lh][n]      = (_Float16)p0;
        Plds[wave][r + 8*lh][16 + n] = (_Float16)p1;
      }

      // P back in A-layout (intra-wave LDS RAW; DS ops in-order)
      v16h pa;
      __builtin_memcpy(&pa,               &Plds[wave][n][16*lh],     16);
      __builtin_memcpy(((char*)&pa) + 16, &Plds[wave][n][16*lh + 8], 16);

      // O += P.V : 4 feature groups + 1 ones-column for the row sums
      #pragma unroll
      for (int g = 0; g < 4; ++g) {
        v16h bv;
        __builtin_memcpy(&bv, &Vt[16*g + n][16*lh], 32);
        acc[g] = __builtin_amdgcn_wmma_f32_16x16x32_f16(false, pa, false, bv, (short)0, acc[g], false, false);
      }
      v8f lsum = __builtin_amdgcn_wmma_f32_16x16x32_f16(false, pa, false, vones, (short)0, zero8, false, false);
      #pragma unroll
      for (int r = 0; r < 8; ++r) l_run[r] += lsum[r];
    } else {
      // ===== TAIL TILE: every score is exactly MASKVAL -> alpha==1, p is a row
      // constant. Accumulate colsum(V) with ones-A WMMAs; fold p in the epilogue.
      #pragma unroll
      for (int g = 0; g < 4; ++g) {
        v16h bv;
        __builtin_memcpy(&bv, &Vt[16*g + n][16*lh], 32);
        ctot[g] = __builtin_amdgcn_wmma_f32_16x16x32_f16(false, vones, false, bv, (short)0, ctot[g], false, false);
      }
    }
  }

  // ---- epilogue: add tail contribution, normalize, store ----
  const float tailkeys = (float)(S_ - KT*(diagkt + 1));
  #pragma unroll
  for (int r = 0; r < 8; ++r) {
    const float pr = exp2f(MASKVAL - m_run[r]);   // tail row constant
    const float l  = l_run[r] + pr * tailkeys;
    const float inv = 1.0f / l;
    #pragma unroll
    for (int g = 0; g < 4; ++g)
      Ob[(size_t)(q0 + r + 8*lh)*D_ + 16*g + n] = (acc[g][r] + pr * ctot[g][r]) * inv;
  }
}

extern "C" void kernel_launch(void* const* d_in, const int* in_sizes, int n_in,
                              void* d_out, int out_size, void* d_ws, size_t ws_size,
                              hipStream_t stream) {
  const float* Q   = (const float*)d_in[0];
  const float* K   = (const float*)d_in[1];
  const float* V   = (const float*)d_in[2];
  const int*   pad = (const int*)d_in[3];
  // d_in[4] = attn_mask (tril) — causality computed analytically in-kernel
  float* out = (float*)d_out;

  dim3 grid(S_ / (WAVES*16), B_ * H_);   // (32, 32)
  dim3 block(WAVES * 32);                // 128 threads = 4 waves
  hipLaunchKernelGGL(fa_wmma_f16_kernel, grid, block, 0, stream, Q, K, V, pad, out);
}
